// MLPRegressor_23656679867724
// MI455X (gfx1250) — matrix operand order repaired and checked by reading, not compiled, not run on hardware
//
#include <hip/hip_runtime.h>
#include <hip/hip_bf16.h>

typedef __attribute__((ext_vector_type(16))) _Float16 v16h;
typedef __attribute__((ext_vector_type(8)))  _Float16 v8h;
typedef __attribute__((ext_vector_type(4)))  _Float16 v4h;
typedef __attribute__((ext_vector_type(8)))  float    v8f;
typedef __attribute__((ext_vector_type(4)))  float    v4f;

#define D_FEAT 64
#define D_EDGE 32
#define IN_DIM 160           // 2*D_FEAT + D_EDGE
#define HID    128
#define TILE_E 128           // edges per tile
#define TPB    4             // edge tiles per block (amortizes W1 staging)
#define ZSTRIDE 176          // Z row stride in halves (352 B: 16B aligned, conflict-skewed)
#define WSTRIDE 168          // W1 row stride in halves (336 B: 16B aligned, conflict-free A frags)

__device__ __forceinline__ v16h concat8(v8h a, v8h b) {
    return __builtin_shufflevector(a, b, 0,1,2,3,4,5,6,7,8,9,10,11,12,13,14,15);
}

__device__ __forceinline__ float leaky(float v) {
    return (v >= 0.0f) ? v : 0.01f * v;
}

__device__ __forceinline__ v4h cvt4(v4f f) {
    v4h hv;
    hv.x = (_Float16)f.x; hv.y = (_Float16)f.y;
    hv.z = (_Float16)f.z; hv.w = (_Float16)f.w;
    return hv;
}

// Gather one 128-edge tile (concat[x[o], x[d], ef]) into LDS as f16.
// 2 threads per edge, 20 x 16B chunks each. Invalid edges -> zeros.
__device__ __forceinline__
void gather_tile(_Float16* __restrict__ zbuf, long long base_e,
                 const float* __restrict__ x, const int* __restrict__ eidx,
                 const float* __restrict__ ef, int n_edges, int n_nodes, int tid) {
    const int e_local  = tid >> 1;      // 0..127
    const int half_sel = tid & 1;       // chunks [0,20) or [20,40)
    const long long eg = base_e + e_local;
    const bool valid = (eg < (long long)n_edges);
    int origin = 0, dest = 0;
    long long eg_c = valid ? eg : 0;
    if (valid) {
        origin = __builtin_nontemporal_load(eidx + eg);
        dest   = __builtin_nontemporal_load(eidx + (long long)n_edges + eg);
        origin = (origin < 0) ? 0 : ((origin >= n_nodes) ? n_nodes - 1 : origin);
        dest   = (dest   < 0) ? 0 : ((dest   >= n_nodes) ? n_nodes - 1 : dest);
    }
    const v4f* sxo = (const v4f*)(x + (size_t)origin * D_FEAT);
    const v4f* sxd = (const v4f*)(x + (size_t)dest   * D_FEAT);
    const v4f* sef = (const v4f*)(ef + (size_t)eg_c * D_EDGE);
    #pragma unroll
    for (int j = 0; j < 20; ++j) {
        const int q = half_sel * 20 + j;      // 0..39 float4 chunks of 160
        v4f f;
        if (!valid)      { f = (v4f)(0.0f); }
        else if (q < 16) { f = sxo[q]; }                               // gather: keep cached (RT)
        else if (q < 32) { f = sxd[q - 16]; }                          // gather: keep cached (RT)
        else             { f = __builtin_nontemporal_load(sef + (q - 32)); } // stream: NT
        *(v4h*)(zbuf + e_local * ZSTRIDE + q * 4) = cvt4(f);  // 8B aligned store
    }
}

// Fused: gather -> f16 staging -> L1 GEMM (WMMA) -> L2 dot -> store
__global__ __launch_bounds__(256)
void edge_mlp_kernel(const float* __restrict__ x,
                     const int*   __restrict__ eidx,   // [2, E]
                     const float* __restrict__ ef,     // [E, 32]
                     const float* __restrict__ W1,     // [128, 160] fp32
                     const float* __restrict__ b1,     // [128]
                     const float* __restrict__ W2,     // [128]
                     const float* __restrict__ b2,     // [1]
                     float* __restrict__ out,
                     int n_edges, int n_nodes) {
    __shared__ _Float16 zsh[2][TILE_E * ZSTRIDE];  // 2 x 45056 B (ping-pong)
    __shared__ _Float16 w1sh[HID * WSTRIDE];       // 43008 B
    __shared__ float w2s[HID];
    __shared__ float b1s[HID];

    const int tid = threadIdx.x;
    const long long block_e0 = (long long)blockIdx.x * (TILE_E * TPB);

    if (tid < HID) { w2s[tid] = W2[tid]; b1s[tid] = b1[tid]; }

    // ---- stage W1 -> LDS as f16, once per block (reused by TPB tiles x 8 waves) ----
    {
        const int row = tid >> 1;           // 0..127
        const int hs  = tid & 1;
        const float* wrow = W1 + (size_t)row * IN_DIM;
        #pragma unroll
        for (int j = 0; j < 20; ++j) {
            const int q = hs * 20 + j;      // 0..39 float4 chunks of 160
            v4f f = *(const v4f*)(wrow + q * 4);
            *(v4h*)(w1sh + row * WSTRIDE + q * 4) = cvt4(f);
        }
    }

    // prologue gather: tile 0 -> buffer 0
    gather_tile(zsh[0], block_e0, x, eidx, ef, n_edges, n_nodes, tid);
    __syncthreads();

    const int lane = tid & 31;
    const int ecol = lane & 15;      // edge-in-tile (B/C column)
    const int hi   = lane >> 4;      // K/M half selector
    const int eb   = (tid >> 5) * 16;
    const float b2v = b2[0];

    for (int t = 0; t < TPB; ++t) {
        // prefetch next tile's gather into the other buffer (overlaps the WMMAs below)
        if (t + 1 < TPB) {
            gather_tile(zsh[(t + 1) & 1], block_e0 + (long long)(t + 1) * TILE_E,
                        x, eidx, ef, n_edges, n_nodes, tid);
        }

        const _Float16* zcur = zsh[t & 1];
        const long long base_e = block_e0 + (long long)t * TILE_E;

        // B fragments: Z^T tiles, reused across all 8 N-tiles.
        // 16-bit B 32x16 layout: two contiguous 8-half runs at k0 and k0+16 (k0 = hi*8), matching A.
        v16h bf[5];
        #pragma unroll
        for (int kt = 0; kt < 5; ++kt) {
            const _Float16* p = zcur + (eb + ecol) * ZSTRIDE + kt * 32 + hi * 8;
            v8h lo = *(const v8h*)p;          // ds_load_b128
            v8h hv = *(const v8h*)(p + 16);   // ds_load_b128
            bf[kt] = concat8(lo, hv);
        }

        float s = 0.0f;
        #pragma unroll
        for (int nt = 0; nt < 8; ++nt) {
            const int row = nt * 16 + ecol;                   // W1 row (A's M index)
            const _Float16* wr = w1sh + row * WSTRIDE;
            v8f c = {};
            #pragma unroll
            for (int kt = 0; kt < 5; ++kt) {
                // 16-bit A 16x32 layout: two contiguous 8-half runs at k0 and k0+16.
                const int k0 = kt * 32 + hi * 8;
                v8h alo = *(const v8h*)(wr + k0);             // ds_load_b128 (conflict-free)
                v8h ahi = *(const v8h*)(wr + k0 + 16);        // ds_load_b128
                v16h a = concat8(alo, ahi);
                c = __builtin_amdgcn_wmma_f32_16x16x32_f16(
                        false, a, false, bf[kt], (short)0, c, false, false);
            }
            // C layout: reg r, this lane -> hidden m = nt*16 + hi*8 + r, edge = ecol.
            const int hbase = nt * 16 + hi * 8;
            #pragma unroll
            for (int r = 0; r < 8; ++r) {
                float h = leaky(c[r] + b1s[hbase + r]);
                s = fmaf(h, w2s[hbase + r], s);
            }
        }

        // lanes l and l+16 hold complementary hidden halves of the same edge
        float tot = s + __shfl_xor(s, 16, 32);
        if (lane < 16) {
            const long long e = base_e + eb + lane;
            if (e < (long long)n_edges) {
                __builtin_nontemporal_store(leaky(tot + b2v), out + e);
            }
        }

        __syncthreads();   // gather(t+1) visible; zsh[t&1] free for gather(t+2)
    }
}

extern "C" void kernel_launch(void* const* d_in, const int* in_sizes, int n_in,
                              void* d_out, int out_size, void* d_ws, size_t ws_size,
                              hipStream_t stream) {
    const float* x   = (const float*)d_in[0];
    const int*   ei  = (const int*)  d_in[1];
    const float* ef  = (const float*)d_in[2];
    const float* W1  = (const float*)d_in[3];
    const float* b1  = (const float*)d_in[4];
    const float* W2  = (const float*)d_in[5];
    const float* b2  = (const float*)d_in[6];
    float* out = (float*)d_out;

    const int n_nodes = in_sizes[0] / D_FEAT;
    const int n_edges = in_sizes[2] / D_EDGE;

    const long long edges_per_block = (long long)TILE_E * TPB;
    const int nblocks = (int)((n_edges + edges_per_block - 1) / edges_per_block);
    edge_mlp_kernel<<<nblocks, 256, 0, stream>>>(x, ei, ef, W1, b1, W2, b2,
                                                 out, n_edges, n_nodes);
}
